// show_attend_tell_2413771620814
// MI455X (gfx1250) — compile-verified, matrix operand order checked by reading, not run on hardware
//
#include <hip/hip_runtime.h>

// ---------- types / helpers ----------
typedef __attribute__((ext_vector_type(16))) __bf16 bf16x16;
typedef __attribute__((ext_vector_type(8)))  float  f32x8;

union FragB16 { bf16x16 v; unsigned int d[8]; uint4 q[2]; };

static __device__ __forceinline__ unsigned short f2bf(float f) {
  union { float f; unsigned u; } x; x.f = f;
  unsigned r = x.u + 0x7FFFu + ((x.u >> 16) & 1u);   // RNE
  return (unsigned short)(r >> 16);
}
static __device__ __forceinline__ float bf2f(unsigned short s) {
  union { unsigned u; float f; } x; x.u = ((unsigned)s) << 16; return x.f;
}
static __device__ __forceinline__ float sigf(float x) { return 1.f / (1.f + expf(-x)); }

// ---------- generic casts ----------
__global__ void cast_f32_bf16(const float* __restrict__ src, unsigned short* __restrict__ dst, long n) {
  for (long i = blockIdx.x * 256L + threadIdx.x; i < n; i += (long)gridDim.x * 256)
    dst[i] = f2bf(src[i]);
}
__global__ void cast_bf16_f32(const unsigned short* __restrict__ src, float* __restrict__ dst, long n) {
  for (long i = blockIdx.x * 256L + threadIdx.x; i < n; i += (long)gridDim.x * 256)
    dst[i] = bf2f(src[i]);
}
// (B,3,224,224) f32 NCHW -> (B,224,224,3) bf16 NHWC
__global__ void img2nhwc(const float* __restrict__ in, unsigned short* __restrict__ out) {
  long n = 8L * 224 * 224 * 3;
  for (long i = blockIdx.x * 256L + threadIdx.x; i < n; i += (long)gridDim.x * 256) {
    int c = (int)(i % 3); long r = i / 3;
    int x = (int)(r % 224); r /= 224;
    int y = (int)(r % 224); int b = (int)(r / 224);
    out[i] = f2bf(in[((long)(b * 3 + c) * 224 + y) * 224 + x]);
  }
}

// ---------- conv as implicit GEMM on WMMA (double-buffered LDS) ----------
// M = 8*H*W (output pixels), N = Cout, K = 9*Cin.  NHWC in/out, HWIO weights (K x N).
#define BM 128
#define BN 64
#define LDP 40   // padded LDS row stride (ushorts): 80B = 5*16B, bank-spreading + 16B aligned
__launch_bounds__(256)
__global__ void conv3x3_wmma(const unsigned short* __restrict__ in,
                             const unsigned short* __restrict__ wgt,
                             const float* __restrict__ bias,
                             unsigned short* __restrict__ out,
                             int H, int W, int Cin, int Cout) {
  __shared__ __align__(16) unsigned short lds_a[2][BM * LDP];    // [buf][row][k]
  __shared__ __align__(16) unsigned short lds_bt[2][BN * LDP];   // [buf][col][k] (transposed)
  const int tid = threadIdx.x;
  const int lane = tid & 31, wave = tid >> 5;
  const int wm = wave & 3, wn = wave >> 2;          // 4x2 wave grid -> 128x64 tile
  const int m0 = blockIdx.y * BM, n0 = blockIdx.x * BN;
  const int M = 8 * H * W, K9 = 9 * Cin, HW = H * W;
  const int half = lane >> 4, r = lane & 15;

  // hoist m -> (b,oy,ox) fully out of the K loop: this thread stages A rows i*8+wave
  int prebase[16]; int preyx[16];
  for (int i = 0; i < 16; ++i) {
    int m = m0 + i * 8 + wave;
    if (m < M) {
      int b = m / HW, rem = m - b * HW;
      int oy = rem / W, ox = rem - oy * W;
      prebase[i] = (b * H + oy) * W + ox;
      preyx[i] = (oy << 16) | ox;
    } else prebase[i] = -1;
  }
  const int bn = tid & 63;           // this thread stages B column bn
  const int bko = tid >> 6;          // and k rows bko, bko+4, ...
  const int cinMask = Cin - 1;
  const int cinShift = (Cin & (Cin - 1)) ? 0 : __builtin_ctz(Cin);  // pow2 for all layers but #1

  auto stageA = [&](int buf, int k0) {
    int k = k0 + lane;
    bool kv = k < K9;
    int dyx = 0, ci = 0, ky = 0, kx = 0;
    if (kv) {
      if (Cin == 3) {                       // layer 1: literal-constant divides
        ky = k / 9; int t2 = k - ky * 9;
        kx = t2 / 3; ci = t2 - kx * 3;
      } else {                              // pow2 Cin: shifts + divide-by-literal-3
        ci = k & cinMask;
        int kxy = k >> cinShift;
        ky = kxy / 3; kx = kxy - ky * 3;
      }
      dyx = (ky - 1) * W + (kx - 1);
    }
    for (int i = 0; i < 16; ++i) {
      unsigned short val = 0;
      if (kv && prebase[i] >= 0) {
        int oy = preyx[i] >> 16, ox = preyx[i] & 0xffff;
        int iy = oy + ky - 1, ix = ox + kx - 1;
        if ((unsigned)iy < (unsigned)H && (unsigned)ix < (unsigned)W)
          val = in[(long)(prebase[i] + dyx) * Cin + ci];
      }
      lds_a[buf][(i * 8 + wave) * LDP + lane] = val;
    }
  };
  auto stageB = [&](int buf, int k0) {
    int nn = n0 + bn;
    for (int i = 0; i < 8; ++i) {
      int kk = i * 4 + bko;
      int k = k0 + kk;
      lds_bt[buf][bn * LDP + kk] =
          (k < K9 && nn < Cout) ? wgt[(long)k * Cout + nn] : (unsigned short)0;
    }
  };

  f32x8 acc[2][2] = {};

  stageA(0, 0); stageB(0, 0);
  __syncthreads();

  int buf = 0;
  for (int k0 = 0; k0 < K9; k0 += 32) {
    // fragments from current buffer: two 16B runs each -> ds_load_b128 x2
    FragB16 afr[2], bfr[2];
    for (int mi = 0; mi < 2; ++mi) {
      int row = wm * 32 + mi * 16 + r;
      afr[mi].q[0] = *(const uint4*)&lds_a[buf][row * LDP + half * 8];
      afr[mi].q[1] = *(const uint4*)&lds_a[buf][row * LDP + 16 + half * 8];
    }
    for (int ni = 0; ni < 2; ++ni) {
      int col = wn * 32 + ni * 16 + r;
      bfr[ni].q[0] = *(const uint4*)&lds_bt[buf][col * LDP + half * 8];
      bfr[ni].q[1] = *(const uint4*)&lds_bt[buf][col * LDP + 16 + half * 8];
    }
    // stage next tile while WMMAs run (global loads overlap matrix pipe)
    if (k0 + 32 < K9) {
      if (k0 + 64 < K9)
        __builtin_prefetch(&wgt[(long)(k0 + 64) * Cout + n0 + bn], 0, 1);
      stageA(buf ^ 1, k0 + 32);
      stageB(buf ^ 1, k0 + 32);
    }
    for (int mi = 0; mi < 2; ++mi)
      for (int ni = 0; ni < 2; ++ni)
        acc[mi][ni] = __builtin_amdgcn_wmma_f32_16x16x32_bf16(
            false, afr[mi].v, false, bfr[ni].v, (short)0, acc[mi][ni], false, false);
    __syncthreads();
    buf ^= 1;
  }
  // bias + ReLU epilogue, bf16 out
  for (int mi = 0; mi < 2; ++mi)
    for (int ni = 0; ni < 2; ++ni)
      for (int v = 0; v < 8; ++v) {
        int m = m0 + wm * 32 + mi * 16 + v + 8 * half;
        int n = n0 + wn * 32 + ni * 16 + r;
        if (m < M && n < Cout) {
          float x = acc[mi][ni][v] + bias[n];
          out[(long)m * Cout + n] = f2bf(x > 0.f ? x : 0.f);
        }
      }
}

__global__ void maxpool2(const unsigned short* __restrict__ in, unsigned short* __restrict__ out,
                         int H, int W, int C) {
  int Ho = H / 2, Wo = W / 2;
  long n = 8L * Ho * Wo * C;
  for (long i = blockIdx.x * 256L + threadIdx.x; i < n; i += (long)gridDim.x * 256) {
    int c = (int)(i % C); long r = i / C;
    int x = (int)(r % Wo); r /= Wo;
    int y = (int)(r % Ho); int b = (int)(r / Ho);
    long base = ((long)(b * H + 2 * y) * W + 2 * x) * C + c;
    unsigned short b00 = in[base], b01 = in[base + C];
    unsigned short b10 = in[base + (long)W * C], b11 = in[base + (long)W * C + C];
    float m = bf2f(b00); unsigned short mb = b00;
    if (bf2f(b01) > m) { m = bf2f(b01); mb = b01; }
    if (bf2f(b10) > m) { m = bf2f(b10); mb = b10; }
    if (bf2f(b11) > m) { mb = b11; }
    out[i] = mb;
  }
}

// ---------- 1-wave WMMA GEMM: C(MxN,f32) = A(MpadxK,bf16) @ W(NxK,bf16)^T (+ bias)(+=) ----------
__launch_bounds__(32)
__global__ void gemm16_wmma(const unsigned short* __restrict__ A,
                            const unsigned short* __restrict__ Wt,
                            const float* __restrict__ bias, float* __restrict__ C,
                            int Mact, int N, int K, int ldc, int accFlag) {
  int lane = threadIdx.x & 31;
  int half = lane >> 4, r = lane & 15;
  int m0 = blockIdx.y * 16, n0 = blockIdx.x * 16;
  f32x8 acc = {};
  if (accFlag) {
    for (int v = 0; v < 8; ++v) {
      int m = m0 + v + 8 * half, n = n0 + r;
      acc[v] = (m < Mact && n < N) ? C[(long)m * ldc + n] : 0.f;
    }
  }
  int n = n0 + r;
  const unsigned short* arow = A + (long)(m0 + r) * K;
  const unsigned short* brow = (n < N) ? (Wt + (long)n * K) : nullptr;
  const uint4 z4 = {0u, 0u, 0u, 0u};
  for (int k0 = 0; k0 < K; k0 += 32) {
    FragB16 a, b;
    a.q[0] = *(const uint4*)(arow + k0 + half * 8);
    a.q[1] = *(const uint4*)(arow + k0 + 16 + half * 8);
    if (brow) {
      b.q[0] = *(const uint4*)(brow + k0 + half * 8);
      b.q[1] = *(const uint4*)(brow + k0 + 16 + half * 8);
    } else { b.q[0] = z4; b.q[1] = z4; }
    acc = __builtin_amdgcn_wmma_f32_16x16x32_bf16(false, a.v, false, b.v, (short)0, acc, false, false);
  }
  for (int v = 0; v < 8; ++v) {
    int m = m0 + v + 8 * half, nn = n0 + r;
    if (m < Mact && nn < N)
      C[(long)m * ldc + nn] = acc[v] + (bias ? bias[nn] : 0.f);
  }
}

// ---------- decoder pointwise kernels ----------
__global__ void avg_feats_kernel(const float* __restrict__ feats, unsigned short* __restrict__ avg_bf) {
  int b = blockIdx.x, d = threadIdx.x;
  float s = 0.f;
  for (int n = 0; n < 196; ++n) s += feats[((long)b * 196 + n) * 512 + d];
  avg_bf[b * 512 + d] = f2bf(s * (1.f / 196.f));
}
__global__ void tanh_init_kernel(float* __restrict__ x, unsigned short* __restrict__ xbf) {
  int b = blockIdx.x, d = threadIdx.x;
  float v = tanhf(x[b * 512 + d]);
  x[b * 512 + d] = v;
  if (xbf) xbf[b * 512 + d] = f2bf(v);
}
__global__ void att_e_kernel(const float* __restrict__ Ws, const float* __restrict__ Uh,
                             const float* __restrict__ vw, const float* __restrict__ vb,
                             float* __restrict__ e) {
  int row = blockIdx.x, b = row / 196, lane = threadIdx.x;
  float s = 0.f;
  for (int d = lane; d < 512; d += 32)
    s += tanhf(Ws[(long)row * 512 + d] + Uh[b * 512 + d]) * vw[d];
  for (int off = 16; off; off >>= 1) s += __shfl_xor(s, off);
  if (lane == 0) e[row] = s + vb[0];
}
__global__ void softmax_alpha(const float* __restrict__ e, float* __restrict__ alpha,
                              float* __restrict__ alphas_out, int t) {
  __shared__ float red[256];
  int b = blockIdx.x, tid = threadIdx.x;
  float v = (tid < 196) ? e[b * 196 + tid] : -3.4e38f;
  red[tid] = v; __syncthreads();
  for (int s = 128; s; s >>= 1) { if (tid < s) red[tid] = fmaxf(red[tid], red[tid + s]); __syncthreads(); }
  float mx = red[0]; __syncthreads();
  float ex = (tid < 196) ? expf(v - mx) : 0.f;
  red[tid] = ex; __syncthreads();
  for (int s = 128; s; s >>= 1) { if (tid < s) red[tid] += red[tid + s]; __syncthreads(); }
  float inv = 1.f / red[0];
  if (tid < 196) {
    float a = ex * inv;
    alpha[b * 196 + tid] = a;
    alphas_out[((long)b * 16 + t) * 196 + tid] = a;
  }
}
__global__ void context_kernel(const float* __restrict__ feats, const float* __restrict__ alpha,
                               float* __restrict__ ctx) {
  int b = blockIdx.x, d = threadIdx.x;
  float s = 0.f;
  for (int n = 0; n < 196; ++n) s += feats[((long)b * 196 + n) * 512 + d] * alpha[b * 196 + n];
  ctx[b * 512 + d] = s;
}
__global__ void lstm_input_kernel(const unsigned short* __restrict__ e_in,
                                  const float* __restrict__ gate_pre,
                                  const float* __restrict__ ctx,
                                  unsigned short* __restrict__ lstm_in) {
  int b = blockIdx.x, j = threadIdx.x;
  unsigned short v;
  if (j < 512) v = e_in[b * 512 + j];
  else {
    int d = j - 512;
    v = f2bf(sigf(gate_pre[b * 512 + d]) * ctx[b * 512 + d]);
  }
  lstm_in[b * 1024 + j] = v;
}
__global__ void lstm_cell_kernel(const float* __restrict__ gates, float* __restrict__ h,
                                 float* __restrict__ c, unsigned short* __restrict__ h_bf) {
  int b = blockIdx.x, d = threadIdx.x;
  const float* g = gates + (long)b * 2048;
  float ig = sigf(g[d]), fg = sigf(g[512 + d]);
  float gg = tanhf(g[1024 + d]), og = sigf(g[1536 + d]);
  float cn = fg * c[b * 512 + d] + ig * gg;
  float hn = og * tanhf(cn);
  c[b * 512 + d] = cn; h[b * 512 + d] = hn; h_bf[b * 512 + d] = f2bf(hn);
}
__global__ void argmax_kernel(const float* __restrict__ preds, int t, int* __restrict__ nxt) {
  __shared__ float sv[256]; __shared__ int si[256];
  int b = blockIdx.x, tid = threadIdx.x;
  const float* row = preds + (long)b * 160000 + (long)t * 10000;
  float best = -3.4e38f; int bi = 0;
  for (int n = tid; n < 10000; n += 256)
    if (row[n] > best) { best = row[n]; bi = n; }
  sv[tid] = best; si[tid] = bi; __syncthreads();
  for (int s = 128; s; s >>= 1) {
    if (tid < s && (sv[tid + s] > sv[tid] ||
                    (sv[tid + s] == sv[tid] && si[tid + s] < si[tid]))) {
      sv[tid] = sv[tid + s]; si[tid] = si[tid + s];
    }
    __syncthreads();
  }
  if (tid == 0) nxt[b] = si[0];
}
__global__ void gather_emb_kernel(const unsigned short* __restrict__ emb_bf,
                                  const int* __restrict__ nxt, unsigned short* __restrict__ e_in) {
  int b = blockIdx.x, d = threadIdx.x;
  e_in[b * 512 + d] = (b < 8) ? emb_bf[(long)nxt[b] * 512 + d] : (unsigned short)0;
}

// ---------- driver ----------
extern "C" void kernel_launch(void* const* d_in, const int* in_sizes, int n_in,
                              void* d_out, int out_size, void* d_ws, size_t ws_size,
                              hipStream_t stream) {
  (void)n_in; (void)out_size; (void)ws_size;
  size_t off = 0;
  auto alloc = [&](size_t bytes) -> void* {
    void* p = (char*)d_ws + off;
    off += (bytes + 255) & ~(size_t)255;
    return p;
  };
  auto cast = [&](const void* src, unsigned short* dst, long n) {
    long g = (n + 1023) / 1024; if (g > 8192) g = 8192; if (g < 1) g = 1;
    cast_f32_bf16<<<dim3((unsigned)g), dim3(256), 0, stream>>>((const float*)src, dst, n);
  };

  // -------- workspace layout --------
  unsigned short* act0 = (unsigned short*)alloc(25690112ull * 2);
  unsigned short* act1 = (unsigned short*)alloc(25690112ull * 2);
  unsigned short* convw[13];
  for (int i = 0; i < 13; ++i) convw[i] = (unsigned short*)alloc((size_t)in_sizes[1 + i] * 2);
  unsigned short* Uw_bf   = (unsigned short*)alloc((size_t)in_sizes[27] * 2);
  unsigned short* Ww_bf   = (unsigned short*)alloc((size_t)in_sizes[29] * 2);
  unsigned short* ihw_bf  = (unsigned short*)alloc((size_t)in_sizes[33] * 2);
  unsigned short* icw_bf  = (unsigned short*)alloc((size_t)in_sizes[35] * 2);
  unsigned short* fbw_bf  = (unsigned short*)alloc((size_t)in_sizes[37] * 2);
  unsigned short* deep_bf = (unsigned short*)alloc((size_t)in_sizes[39] * 2);
  unsigned short* emb_bf  = (unsigned short*)alloc((size_t)in_sizes[41] * 2);
  unsigned short* wih_bf  = (unsigned short*)alloc((size_t)in_sizes[42] * 2);
  unsigned short* whh_bf  = (unsigned short*)alloc((size_t)in_sizes[43] * 2);
  float* feats = (float*)alloc(1568ull * 512 * 4);
  float* Ws    = (float*)alloc(1568ull * 512 * 4);
  float* Uh    = (float*)alloc(16ull * 512 * 4);
  float* tmp   = (float*)alloc(16ull * 512 * 4);
  float* e     = (float*)alloc(1568ull * 4);
  float* alpha = (float*)alloc(1568ull * 4);
  float* ctx   = (float*)alloc(8ull * 512 * 4);
  float* h     = (float*)alloc(16ull * 512 * 4);
  float* c     = (float*)alloc(16ull * 512 * 4);
  float* gates = (float*)alloc(16ull * 2048 * 4);
  unsigned short* h_bf    = (unsigned short*)alloc(16ull * 512 * 2);
  unsigned short* avg_bf  = (unsigned short*)alloc(16ull * 512 * 2);
  unsigned short* e_in    = (unsigned short*)alloc(16ull * 512 * 2);
  unsigned short* lstm_in = (unsigned short*)alloc(16ull * 1024 * 2);
  int* nxt = (int*)alloc(8 * 4);

  float* preds      = (float*)d_out;
  float* alphas_out = (float*)d_out + 8ull * 16 * 10000;

  // -------- weight casts (f32 -> bf16), inside graph --------
  for (int i = 0; i < 13; ++i) cast(d_in[1 + i], convw[i], in_sizes[1 + i]);
  cast(d_in[27], Uw_bf, in_sizes[27]);
  cast(d_in[29], Ww_bf, in_sizes[29]);
  cast(d_in[33], ihw_bf, in_sizes[33]);
  cast(d_in[35], icw_bf, in_sizes[35]);
  cast(d_in[37], fbw_bf, in_sizes[37]);
  cast(d_in[39], deep_bf, in_sizes[39]);
  cast(d_in[41], emb_bf, in_sizes[41]);
  cast(d_in[42], wih_bf, in_sizes[42]);
  cast(d_in[43], whh_bf, in_sizes[43]);

  // zero padded rows (rows 8..15 of M=16-padded GEMM A operands)
  hipMemsetAsync(h_bf, 0, 16 * 512 * 2, stream);
  hipMemsetAsync(avg_bf, 0, 16 * 512 * 2, stream);
  hipMemsetAsync(lstm_in, 0, 16 * 1024 * 2, stream);
  hipMemsetAsync(nxt, 0, 8 * 4, stream);

  // -------- VGG16 feature stack (bf16 WMMA implicit GEMM) --------
  img2nhwc<<<dim3(4704), dim3(256), 0, stream>>>((const float*)d_in[0], act0);
  const int cfg[17] = {64, 64, -1, 128, 128, -1, 256, 256, 256, -1,
                       512, 512, 512, -1, 512, 512, 512};
  int H = 224, Wd = 224, cin = 3, wi = 0;
  unsigned short* cur = act0; unsigned short* oth = act1;
  for (int ci = 0; ci < 17; ++ci) {
    int cc = cfg[ci];
    if (cc < 0) {
      int Ho = H / 2, Wo = Wd / 2;
      long n = 8L * Ho * Wo * cin;
      long g = (n + 255) / 256; if (g > 32768) g = 32768;
      maxpool2<<<dim3((unsigned)g), dim3(256), 0, stream>>>(cur, oth, H, Wd, cin);
      H = Ho; Wd = Wo;
    } else {
      int M = 8 * H * Wd;
      dim3 g((cc + BN - 1) / BN, (M + BM - 1) / BM);
      conv3x3_wmma<<<g, dim3(256), 0, stream>>>(cur, convw[wi], (const float*)d_in[14 + wi],
                                                oth, H, Wd, cin, cc);
      cin = cc; ++wi;
    }
    unsigned short* t = cur; cur = oth; oth = t;
  }
  unsigned short* feats_bf = cur;   // (8,14,14,512) == (1568, 512) bf16

  // feats f32, mean, LSTM init
  cast_bf16_f32<<<dim3(3136), dim3(256), 0, stream>>>(feats_bf, feats, 1568L * 512);
  avg_feats_kernel<<<dim3(8), dim3(512), 0, stream>>>(feats, avg_bf);
  gemm16_wmma<<<dim3(32, 1), dim3(32), 0, stream>>>(avg_bf, ihw_bf, (const float*)d_in[34], h,
                                                    8, 512, 512, 512, 0);
  gemm16_wmma<<<dim3(32, 1), dim3(32), 0, stream>>>(avg_bf, icw_bf, (const float*)d_in[36], c,
                                                    8, 512, 512, 512, 0);
  tanh_init_kernel<<<dim3(8), dim3(512), 0, stream>>>(h, h_bf);
  tanh_init_kernel<<<dim3(8), dim3(512), 0, stream>>>(c, (unsigned short*)nullptr);
  gather_emb_kernel<<<dim3(16), dim3(512), 0, stream>>>(emb_bf, nxt, e_in); // word 0

  // hoisted (loop-invariant): W_s = feats @ W_w^T + W_b   (1568 x 512)
  gemm16_wmma<<<dim3(32, 98), dim3(32), 0, stream>>>(feats_bf, Ww_bf, (const float*)d_in[30],
                                                     Ws, 1568, 512, 512, 512, 0);

  // -------- 16 decode steps --------
  for (int t = 0; t < 16; ++t) {
    gemm16_wmma<<<dim3(32, 1), dim3(32), 0, stream>>>(h_bf, Uw_bf, (const float*)d_in[28],
                                                      Uh, 8, 512, 512, 512, 0);
    att_e_kernel<<<dim3(1568), dim3(32), 0, stream>>>(Ws, Uh, (const float*)d_in[31],
                                                      (const float*)d_in[32], e);
    softmax_alpha<<<dim3(8), dim3(256), 0, stream>>>(e, alpha, alphas_out, t);
    context_kernel<<<dim3(8), dim3(512), 0, stream>>>(feats, alpha, ctx);
    gemm16_wmma<<<dim3(32, 1), dim3(32), 0, stream>>>(h_bf, fbw_bf, (const float*)d_in[38],
                                                      tmp, 8, 512, 512, 512, 0);
    lstm_input_kernel<<<dim3(8), dim3(1024), 0, stream>>>(e_in, tmp, ctx, lstm_in);
    gemm16_wmma<<<dim3(128, 1), dim3(32), 0, stream>>>(lstm_in, wih_bf, (const float*)d_in[44],
                                                       gates, 8, 2048, 1024, 2048, 0);
    gemm16_wmma<<<dim3(128, 1), dim3(32), 0, stream>>>(h_bf, whh_bf, (const float*)d_in[45],
                                                       gates, 8, 2048, 512, 2048, 1);
    lstm_cell_kernel<<<dim3(8), dim3(512), 0, stream>>>(gates, h, c, h_bf);
    // classifier straight into preds[b][t][:]
    gemm16_wmma<<<dim3(625, 1), dim3(32), 0, stream>>>(h_bf, deep_bf, (const float*)d_in[40],
                                                       preds + (long)t * 10000,
                                                       8, 10000, 512, 160000, 0);
    argmax_kernel<<<dim3(8), dim3(256), 0, stream>>>(preds, t, nxt);
    gather_emb_kernel<<<dim3(16), dim3(512), 0, stream>>>(emb_bf, nxt, e_in);
  }
}
